// Multi_Head_Attention_35708358099228
// MI455X (gfx1250) — matrix operand order repaired and checked by reading, compiled, not run on hardware
//
#include <hip/hip_runtime.h>

// ---------------------------------------------------------------------------
// Multi-head attention, CDNA5 (gfx1250), wave32, WMMA f32_16x16x32_f16.
// B=4, N=2048, D=64 (head dim), H=8, DH = D*H = 512.
// Weights are pre-packed once into WMMA B-fragment order (f16) so all hot
// loops issue only contiguous b128 loads.
// ---------------------------------------------------------------------------

typedef _Float16 half_t;
typedef __attribute__((ext_vector_type(16))) _Float16 v16h;
typedef __attribute__((ext_vector_type(8)))  _Float16 v8h;
typedef __attribute__((ext_vector_type(8)))  float    v8f;
typedef __attribute__((ext_vector_type(4)))  float    v4f;

#define WMMA_F16(A, Bm, C) \
    __builtin_amdgcn_wmma_f32_16x16x32_f16(false, (A), false, (Bm), (short)0, (C), false, false)

static __device__ __forceinline__ v16h cat8(v8h lo, v8h hi) {
    return __builtin_shufflevector(lo, hi, 0, 1, 2, 3, 4, 5, 6, 7,
                                           8, 9, 10, 11, 12, 13, 14, 15);
}

static __device__ __forceinline__ v16h load_frag16(const half_t* __restrict__ p) {
    const v8h* __restrict__ p8 = (const v8h*)p;
    return cat8(p8[0], p8[1]);
}

#define BATCH 4
#define SEQ   2048
#define HDIM  64
#define HEADS 8
#define DH    512   // HDIM*HEADS

// Packed-weight geometry:
//  WqkvP: fragment f = mat*64 + kc*32 + ct  (mat 0..2, kc 0..1, ct 0..31)
//         layout [f][lane(32)][i(16)] halves -> 192 frags * 512 = 98304 halves
//  WuP:   fragment f = kc*4 + ct            (kc 0..15, ct 0..3)
//         layout [f][lane(32)][i(16)] halves ->  64 frags * 512 = 32768 halves
#define QKV_FRAGS 192
#define WU_FRAGS  64

// ---------------------------------------------------------------------------
// Kernel 0: one-time weight pack (f32 column gathers -> f16 fragment order).
// grid: QKV_FRAGS + WU_FRAGS = 256 blocks, 32 threads (1 wave)
// ---------------------------------------------------------------------------
__global__ void pack_w_kernel(const float* __restrict__ Wq,
                              const float* __restrict__ Wk,
                              const float* __restrict__ Wv,
                              const float* __restrict__ Wu,
                              half_t* __restrict__ WqkvP,
                              half_t* __restrict__ WuP) {
    const int lane = threadIdx.x;
    const int n  = lane & 15;
    const int hi = lane >> 4;
    const int idx = blockIdx.x;

    v8h lo, hv;
    if (idx < QKV_FRAGS) {
        const int kc  = (idx >> 5) & 1;
        const int ct  = idx & 31;
        const int mat = idx >> 6;         // /64
        const float* __restrict__ W = (mat == 0) ? Wq : (mat == 1) ? Wk : Wv;
#pragma unroll
        for (int i = 0; i < 8; ++i) {
            lo[i] = (half_t)W[(size_t)(kc * 32 + 8 * hi + i)     * DH + ct * 16 + n];
            hv[i] = (half_t)W[(size_t)(kc * 32 + 16 + 8 * hi + i) * DH + ct * 16 + n];
        }
        half_t* __restrict__ dst = WqkvP + ((size_t)idx * 32 + lane) * 16;
        ((v8h*)dst)[0] = lo;
        ((v8h*)dst)[1] = hv;
    } else {
        const int f  = idx - QKV_FRAGS;   // 0..63
        const int kc = f >> 2;
        const int ct = f & 3;
#pragma unroll
        for (int i = 0; i < 8; ++i) {
            lo[i] = (half_t)Wu[(size_t)(kc * 32 + 8 * hi + i)     * HDIM + ct * 16 + n];
            hv[i] = (half_t)Wu[(size_t)(kc * 32 + 16 + 8 * hi + i) * HDIM + ct * 16 + n];
        }
        half_t* __restrict__ dst = WuP + ((size_t)f * 32 + lane) * 16;
        ((v8h*)dst)[0] = lo;
        ((v8h*)dst)[1] = hv;
    }
}

// ---------------------------------------------------------------------------
// Kernel 1: QKV projections from packed weights.
//   One wave per (16-row tile, head); loops Q,K,V so x A-fragments are loaded
//   once and feed 24 WMMAs. sched_barrier per tile bounds live registers
//   (prevents the scheduler from pipelining all 12 tiles -> scratch spills).
//   Q,K stored [b,h,n,64]; V stored [b,h,64,n].
// grid: (B*N/16, H), block: 32 threads (1 wave)
// ---------------------------------------------------------------------------
__global__ void proj_kernel(const float* __restrict__ x,
                            const half_t* __restrict__ WqkvP,
                            half_t* __restrict__ Qh,
                            half_t* __restrict__ Kh,
                            half_t* __restrict__ Vt) {
    const int lane = threadIdx.x;
    const int m  = lane & 15;
    const int hi = lane >> 4;
    const int tile = blockIdx.x;
    const int h    = blockIdx.y;

    const int rowbase = tile * 16;           // never crosses a batch boundary
    const int arow    = rowbase + m;
    const int bb      = rowbase >> 11;       // / SEQ (block-uniform)
    const int nbase   = rowbase & (SEQ - 1);
    const size_t headIdx = (size_t)(bb * HEADS + h);

    // A fragments (f32 -> f16): two K-chunks of 32 over d = 0..63; b128 loads.
    const v4f* __restrict__ x4 = (const v4f*)(x + (size_t)arow * HDIM);
    const v4f c0 = x4[2 * hi],      c1 = x4[2 * hi + 1];
    const v4f c2 = x4[4 + 2 * hi],  c3 = x4[5 + 2 * hi];
    const v4f c4 = x4[8 + 2 * hi],  c5 = x4[9 + 2 * hi];
    const v4f c6 = x4[12 + 2 * hi], c7 = x4[13 + 2 * hi];
    v16h a0, a1;
#pragma unroll
    for (int i = 0; i < 4; ++i) {
        a0[i]      = (half_t)c0[i];  a0[i + 4]  = (half_t)c1[i];
        a0[i + 8]  = (half_t)c2[i];  a0[i + 12] = (half_t)c3[i];
        a1[i]      = (half_t)c4[i];  a1[i + 4]  = (half_t)c5[i];
        a1[i + 8]  = (half_t)c6[i];  a1[i + 12] = (half_t)c7[i];
    }

    half_t* __restrict__ qDst =
        Qh + (headIdx * SEQ + (size_t)(nbase + 8 * hi)) * HDIM;
    half_t* __restrict__ kDst =
        Kh + (headIdx * SEQ + (size_t)(nbase + 8 * hi)) * HDIM;

#pragma unroll
    for (int mat = 0; mat < 3; ++mat) {
#pragma unroll
        for (int dt = 0; dt < 4; ++dt) {
            const int ct = h * 4 + dt;  // global 16-col tile (0..31)
            const v16h b0 = load_frag16(
                WqkvP + ((size_t)(mat * 64 +      ct) * 32 + lane) * 16);
            const v16h b1 = load_frag16(
                WqkvP + ((size_t)(mat * 64 + 32 + ct) * 32 + lane) * 16);

            v8f c;
#pragma unroll
            for (int r = 0; r < 8; ++r) c[r] = 0.0f;
            c = WMMA_F16(a0, b0, c);
            c = WMMA_F16(a1, b1, c);

            if (mat == 2) {
                // V transposed: 8 C-rows per lane = 8 contiguous n -> b128.
                v8h pack;
#pragma unroll
                for (int r = 0; r < 8; ++r) pack[r] = (half_t)c[r];
                *(v8h*)(Vt + (headIdx * HDIM + (size_t)(dt * 16 + m)) * SEQ +
                        nbase + 8 * hi) = pack;
            } else {
                half_t* __restrict__ dst =
                    ((mat == 0) ? qDst : kDst) + dt * 16 + m;
#pragma unroll
                for (int r = 0; r < 8; ++r)
                    dst[(size_t)r * HDIM] = (half_t)c[r];
            }
            // Keep tiles in program order: bounds live VGPRs, avoids spills.
            __builtin_amdgcn_sched_barrier(0);
        }
    }
}

// ---------------------------------------------------------------------------
// Kernel 2: flash attention, one wave per (b, h, 32-query block).
// Two 16-row q-tiles share every K/V fragment; online softmax uses all 32
// lanes (row = lane). Score/P tiles staged through LDS.
// grid: (N/32, H, B), block: 32 threads (1 wave)
// ---------------------------------------------------------------------------
__global__ void attn_kernel(const half_t* __restrict__ Qh,
                            const half_t* __restrict__ Kh,
                            const half_t* __restrict__ Vt,
                            half_t* __restrict__ Y) {
    __shared__ float  s_scores[32][32];
    __shared__ half_t s_p[32][32];
    __shared__ float  s_scale[32];
    __shared__ float  s_inv[32];

    const int lane = threadIdx.x;
    const int m  = lane & 15;
    const int hi = lane >> 4;
    const int qb = blockIdx.x;
    const int h  = blockIdx.y;
    const int b  = blockIdx.z;

    const half_t* __restrict__ Qp = Qh + (size_t)(b * HEADS + h) * SEQ * HDIM;
    const half_t* __restrict__ Kp = Kh + (size_t)(b * HEADS + h) * SEQ * HDIM;
    const half_t* __restrict__ Vp = Vt + (size_t)(b * HEADS + h) * HDIM * SEQ;

    const int q0 = qb * 32;

    v16h qa[2][2];
#pragma unroll
    for (int qt = 0; qt < 2; ++qt) {
        const v8h* __restrict__ q8 =
            (const v8h*)(Qp + (size_t)(q0 + qt * 16 + m) * HDIM);
        qa[qt][0] = cat8(q8[hi],     q8[2 + hi]);
        qa[qt][1] = cat8(q8[4 + hi], q8[6 + hi]);
    }

    v8f acc[2][4];
#pragma unroll
    for (int qt = 0; qt < 2; ++qt)
#pragma unroll
        for (int dt = 0; dt < 4; ++dt)
#pragma unroll
            for (int r = 0; r < 8; ++r) acc[qt][dt][r] = 0.0f;

    float rmax = -__builtin_inff();  // per-lane row state, row = lane
    float rsum = 0.0f;

    for (int k0 = 0; k0 < SEQ; k0 += 32) {
        // ---- scores: 2 key-tiles x 2 q-tiles ------------------------------
#pragma unroll
        for (int kt = 0; kt < 2; ++kt) {
            const v8h* __restrict__ k8 =
                (const v8h*)(Kp + (size_t)(k0 + kt * 16 + m) * HDIM);
            const v16h kb0 = cat8(k8[hi],     k8[2 + hi]);
            const v16h kb1 = cat8(k8[4 + hi], k8[6 + hi]);
#pragma unroll
            for (int qt = 0; qt < 2; ++qt) {
                v8f s;
#pragma unroll
                for (int r = 0; r < 8; ++r) s[r] = 0.0f;
                s = WMMA_F16(qa[qt][0], kb0, s);
                s = WMMA_F16(qa[qt][1], kb1, s);
#pragma unroll
                for (int r = 0; r < 8; ++r)
                    s_scores[qt * 16 + r + 8 * hi][kt * 16 + m] =
                        s[r] * 0.125f;  // 1/sqrt(64)
            }
        }
        __syncthreads();

        // ---- online softmax, all 32 lanes (row = lane) --------------------
        {
            float cmax = -__builtin_inff();
#pragma unroll
            for (int j = 0; j < 32; ++j) cmax = fmaxf(cmax, s_scores[lane][j]);
            const float mnew = fmaxf(rmax, cmax);
            const float corr = __expf(rmax - mnew);  // first iter: exp(-inf)=0
            float sum = 0.0f;
#pragma unroll
            for (int j = 0; j < 32; ++j) {
                const float e = __expf(s_scores[lane][j] - mnew);
                sum += e;
                s_p[lane][j] = (half_t)e;
            }
            rsum = rsum * corr + sum;
            rmax = mnew;
            s_scale[lane] = corr;
        }
        __syncthreads();

        // ---- rescale accumulators -----------------------------------------
#pragma unroll
        for (int qt = 0; qt < 2; ++qt)
#pragma unroll
            for (int r = 0; r < 8; ++r) {
                const float sc = s_scale[qt * 16 + r + 8 * hi];
                acc[qt][0][r] *= sc; acc[qt][1][r] *= sc;
                acc[qt][2][r] *= sc; acc[qt][3][r] *= sc;
            }

        // ---- P as A-fragments from LDS (ds_load_b128) ---------------------
        v16h pa[2];
#pragma unroll
        for (int qt = 0; qt < 2; ++qt)
            pa[qt] = cat8(*(const v8h*)&s_p[qt * 16 + m][16 * hi],
                          *(const v8h*)&s_p[qt * 16 + m][16 * hi + 8]);

        // ---- PV: each V fragment feeds both q-tiles -----------------------
#pragma unroll
        for (int dt = 0; dt < 4; ++dt) {
            const half_t* __restrict__ vptr =
                Vp + (size_t)(dt * 16 + m) * SEQ + k0 + 16 * hi;
            const v16h vb = cat8(*(const v8h*)vptr, *(const v8h*)(vptr + 8));
            acc[0][dt] = WMMA_F16(pa[0], vb, acc[0][dt]);
            acc[1][dt] = WMMA_F16(pa[1], vb, acc[1][dt]);
        }
        __syncthreads();
    }

    s_inv[lane] = 1.0f / rsum;
    __syncthreads();

    // ---- write Y [b, n, h*64 + d] as f16 ----------------------------------
#pragma unroll
    for (int qt = 0; qt < 2; ++qt)
#pragma unroll
        for (int dt = 0; dt < 4; ++dt)
#pragma unroll
            for (int r = 0; r < 8; ++r) {
                const int q   = q0 + qt * 16 + r + 8 * hi;
                const float v = acc[qt][dt][r] * s_inv[qt * 16 + r + 8 * hi];
                Y[((size_t)(b * SEQ + q)) * DH + h * HDIM + dt * 16 + m] =
                    (half_t)v;
            }
}

// ---------------------------------------------------------------------------
// Kernel 3: output projection from packed Wu.
//   Y[8192,512] f16 @ Wu[512,64] + bu -> f32 out.
// grid: (B*N/16), block: 32 threads (1 wave); K=512 -> 16-chunk WMMA chain.
// ---------------------------------------------------------------------------
__global__ void outproj_kernel(const half_t* __restrict__ Y,
                               const half_t* __restrict__ WuP,
                               const float* __restrict__ bu,
                               float* __restrict__ out) {
    const int lane = threadIdx.x;
    const int m  = lane & 15;
    const int hi = lane >> 4;
    const int rowbase = blockIdx.x * 16;
    const int arow    = rowbase + m;

    v8f acc[4];
#pragma unroll
    for (int ct = 0; ct < 4; ++ct)
#pragma unroll
        for (int r = 0; r < 8; ++r) acc[ct][r] = 0.0f;

    const v8h* __restrict__ y8 = (const v8h*)(Y + (size_t)arow * DH);

#pragma unroll 4
    for (int kc = 0; kc < 16; ++kc) {  // 16 chunks of 32 over K=512
        const v16h a = cat8(y8[kc * 4 + hi], y8[kc * 4 + 2 + hi]);
#pragma unroll
        for (int ct = 0; ct < 4; ++ct) {
            const v16h bf = load_frag16(
                WuP + ((size_t)(kc * 4 + ct) * 32 + lane) * 16);
            acc[ct] = WMMA_F16(a, bf, acc[ct]);
        }
    }

#pragma unroll
    for (int ct = 0; ct < 4; ++ct) {
        const float bias = bu[ct * 16 + m];
#pragma unroll
        for (int r = 0; r < 8; ++r) {
            const int orow = rowbase + r + 8 * hi;
            out[(size_t)orow * HDIM + ct * 16 + m] = acc[ct][r] + bias;
        }
    }
}

// ---------------------------------------------------------------------------
// Launch
// ---------------------------------------------------------------------------
extern "C" void kernel_launch(void* const* d_in, const int* in_sizes, int n_in,
                              void* d_out, int out_size, void* d_ws, size_t ws_size,
                              hipStream_t stream) {
    const float* x  = (const float*)d_in[0];
    const float* Wq = (const float*)d_in[1];
    const float* Wk = (const float*)d_in[2];
    const float* Wv = (const float*)d_in[3];
    const float* Wu = (const float*)d_in[4];
    const float* bu = (const float*)d_in[5];
    float* out = (float*)d_out;

    const size_t headElems = (size_t)BATCH * HEADS * SEQ * HDIM;  // 4,194,304
    half_t* Qh    = (half_t*)d_ws;
    half_t* Kh    = Qh + headElems;
    half_t* Vt    = Kh + headElems;
    half_t* Yb    = Vt + headElems;              // [B*N, 512] f16
    half_t* WqkvP = Yb + (size_t)BATCH * SEQ * DH;
    half_t* WuP   = WqkvP + (size_t)QKV_FRAGS * 512;
    // total ws use: 32 MB + 256 KB

    pack_w_kernel<<<dim3(QKV_FRAGS + WU_FRAGS), 32, 0, stream>>>(
        Wq, Wk, Wv, Wu, WqkvP, WuP);

    dim3 gProj(BATCH * SEQ / 16, HEADS);         // 512 x 8 waves
    proj_kernel<<<gProj, 32, 0, stream>>>(x, WqkvP, Qh, Kh, Vt);

    dim3 gAttn(SEQ / 32, HEADS, BATCH);          // 64 x 8 x 4 waves
    attn_kernel<<<gAttn, 32, 0, stream>>>(Qh, Kh, Vt, Yb);

    outproj_kernel<<<dim3(BATCH * SEQ / 16), 32, 0, stream>>>(Yb, WuP, bu, out);
}
